// BasicLSTM_46162308497954
// MI455X (gfx1250) — matrix operand order repaired and checked by reading, not compiled, not run on hardware
//
#include <hip/hip_runtime.h>
#include <hip/hip_bf16.h>

typedef __attribute__((ext_vector_type(16))) __bf16 v16bf;
typedef __attribute__((ext_vector_type(8)))  __bf16 v8bf;
typedef __attribute__((ext_vector_type(8)))  float  v8f;

union F16Frag { v16bf v; v8bf h[2]; };

__device__ __forceinline__ float sigmoidf_(float x) {
  return 1.0f / (1.0f + __expf(-x));
}

// ---------------- prep kernels ----------------

__global__ void cvt_f32_to_bf16_k(const float* __restrict__ src,
                                  __bf16* __restrict__ dst, int n) {
  int i = blockIdx.x * blockDim.x + threadIdx.x;
  if (i < n) dst[i] = (__bf16)src[i];
}

// src: [K][N] fp32 (row-major)  ->  dst: [N][K] bf16 (row-major, i.e. transposed)
__global__ void transpose_f32_to_bf16_k(const float* __restrict__ src,
                                        __bf16* __restrict__ dst, int K, int N) {
  int i = blockIdx.x * blockDim.x + threadIdx.x;
  if (i < K * N) {
    int n = i / K;
    int k = i - n * K;
    dst[i] = (__bf16)src[(size_t)k * N + n];
  }
}

__global__ void init_state_k(__bf16* __restrict__ hbuf, unsigned* __restrict__ bar, int nh) {
  int i = blockIdx.x * blockDim.x + threadIdx.x;
  if (i < nh) hbuf[i] = (__bf16)0.0f;
  if (i < 2) bar[i] = 0u;
}

// ---------------- phase 1: xz = x @ W + b  (M=16384, K=512, N=4096) ----------------
// 1 wave = 1 M-tile x 4 N-tiles (A-fragment reuse).  65536 waves total.
__global__ __launch_bounds__(256) void lstm_xproj_wmma(
    const __bf16* __restrict__ xb,   // [16384][512] bf16
    const __bf16* __restrict__ Wt,   // [4096][512]  bf16 (W transposed)
    const float*  __restrict__ bias, // [4096]
    float*        __restrict__ xz)   // [T=512][B=32][4096] fp32
{
  const int lane = threadIdx.x & 31;
  const int wave = blockIdx.x * 8 + (threadIdx.x >> 5);
  const int mt = wave >> 6;   // 0..1023  (M tile)
  const int ng = wave & 63;   // 0..63    (group of 4 N tiles)
  const int lm = lane & 15;
  const int lh = lane >> 4;

  const __bf16* arow = xb + (size_t)(mt * 16 + lm) * 512;

  v8f acc[4] = {};

  for (int k0 = 0; k0 < 512; k0 += 32) {
    F16Frag a;
    a.h[0] = *(const v8bf*)(arow + k0 + lh * 8);
    a.h[1] = *(const v8bf*)(arow + k0 + 16 + lh * 8);
    if (k0 + 32 < 512) __builtin_prefetch(arow + k0 + 32, 0, 0);
#pragma unroll
    for (int j = 0; j < 4; ++j) {
      const int n = (ng * 4 + j) * 16 + lm;
      const __bf16* brow = Wt + (size_t)n * 512 + k0;
      F16Frag bm;
      bm.h[0] = *(const v8bf*)(brow + lh * 8);
      bm.h[1] = *(const v8bf*)(brow + 16 + lh * 8);
      acc[j] = __builtin_amdgcn_wmma_f32_16x16x32_bf16(
          false, a.v, false, bm.v, (short)0, acc[j], false, false);
    }
  }

  // store with bias; big-M row r = b*512 + t  ->  xz[(t*32+b)*4096 + n]
#pragma unroll
  for (int j = 0; j < 4; ++j) {
    const int n = (ng * 4 + j) * 16 + lm;
    const float bv = bias[n];
#pragma unroll
    for (int r = 0; r < 8; ++r) {
      const int row = mt * 16 + lh * 8 + r;
      const int bi = row >> 9;
      const int t  = row & 511;
      xz[((size_t)t * 32 + bi) * 4096 + n] = acc[j][r] + bv;
    }
  }
}

// ---------------- device-wide barrier (persistent kernel, 16 blocks) ----------------
__device__ __forceinline__ void grid_barrier(unsigned* count, unsigned* gen, int nblocks) {
  __syncthreads();
  if (threadIdx.x == 0) {
    __threadfence();
    const unsigned my = __hip_atomic_load(gen, __ATOMIC_RELAXED, __HIP_MEMORY_SCOPE_AGENT);
    const unsigned prev = __hip_atomic_fetch_add(count, 1u, __ATOMIC_ACQ_REL, __HIP_MEMORY_SCOPE_AGENT);
    if (prev == (unsigned)nblocks - 1u) {
      __hip_atomic_store(count, 0u, __ATOMIC_RELAXED, __HIP_MEMORY_SCOPE_AGENT);
      __hip_atomic_fetch_add(gen, 1u, __ATOMIC_RELEASE, __HIP_MEMORY_SCOPE_AGENT);
    } else {
      while (__hip_atomic_load(gen, __ATOMIC_ACQUIRE, __HIP_MEMORY_SCOPE_AGENT) == my) {
        __builtin_amdgcn_s_sleep(2);
      }
    }
    __threadfence();
  }
  __syncthreads();
}

// ---------------- phase 2: persistent recurrent kernel ----------------
// 16 blocks x 8 waves = 128 waves.  wave -> (mt in 0..1 batch tile, ut in 0..63 unit tile).
// Each wave computes z tiles for all 4 gates of its (batch-tile, unit-tile), so the
// cell update is wave-local; c lives in registers across all 512 steps.
__global__ __launch_bounds__(256) void lstm_recurrent_wmma(
    const float*  __restrict__ xz,   // [512][32][4096] fp32 (bias already folded in)
    const __bf16* __restrict__ Ut,   // [4096][1024] bf16 (U transposed)
    __bf16*       __restrict__ hbuf, // 2 x [32][1024] bf16, ping-pong
    float*        __restrict__ out,  // [32][512][1024] fp32
    unsigned*     __restrict__ bar, int nblocks)
{
  const int lane = threadIdx.x & 31;
  const int wave = blockIdx.x * 8 + (threadIdx.x >> 5); // 0..127
  const int mt = wave >> 6;   // batch tile 0..1
  const int ut = wave & 63;   // unit tile 0..63
  const int lm = lane & 15;
  const int lh = lane >> 4;
  const int u  = ut * 16 + lm;

  v8f c = {};

  for (int t = 0; t < 512; ++t) {
    const __bf16* hread  = hbuf + (size_t)(t & 1) * (32 * 1024);
    __bf16*       hwrite = hbuf + (size_t)((t + 1) & 1) * (32 * 1024);
    const float*  xzt    = xz + (size_t)t * 32 * 4096;

    // accumulators start from the precomputed input projection (+bias)
    v8f acc[4];
#pragma unroll
    for (int g = 0; g < 4; ++g) {
      const int n = g * 1024 + u;
#pragma unroll
      for (int r = 0; r < 8; ++r) {
        const int row = mt * 16 + lh * 8 + r;
        acc[g][r] = xzt[(size_t)row * 4096 + n];
      }
    }

    const __bf16* arow = hread + (size_t)(mt * 16 + lm) * 1024;
    for (int k0 = 0; k0 < 1024; k0 += 32) {
      F16Frag a;
      a.h[0] = *(const v8bf*)(arow + k0 + lh * 8);
      a.h[1] = *(const v8bf*)(arow + k0 + 16 + lh * 8);
#pragma unroll
      for (int g = 0; g < 4; ++g) {
        const int n = g * 1024 + u;
        const __bf16* brow = Ut + (size_t)n * 1024 + k0;
        F16Frag bm;
        bm.h[0] = *(const v8bf*)(brow + lh * 8);
        bm.h[1] = *(const v8bf*)(brow + 16 + lh * 8);
        acc[g] = __builtin_amdgcn_wmma_f32_16x16x32_bf16(
            false, a.v, false, bm.v, (short)0, acc[g], false, false);
      }
    }

    // gates + state update (fp32), write h to output and bf16 ping-pong buffer
#pragma unroll
    for (int r = 0; r < 8; ++r) {
      const float ig = sigmoidf_(acc[0][r]);
      const float fg = sigmoidf_(acc[1][r]);
      const float gg = tanhf(acc[2][r]);
      const float og = sigmoidf_(acc[3][r]);
      const float cn = fg * c[r] + ig * gg;
      c[r] = cn;
      const float hn = og * tanhf(cn);
      const int row = mt * 16 + lh * 8 + r;  // batch index
      out[((size_t)row * 512 + t) * 1024 + u] = hn;
      hwrite[(size_t)row * 1024 + u] = (__bf16)hn;
    }

    grid_barrier(bar, bar + 1, nblocks);
  }
}

// ---------------- launch ----------------

extern "C" void kernel_launch(void* const* d_in, const int* in_sizes, int n_in,
                              void* d_out, int out_size, void* d_ws, size_t ws_size,
                              hipStream_t stream) {
  (void)in_sizes; (void)n_in; (void)out_size; (void)ws_size;
  const float* x = (const float*)d_in[0];  // [32][512][512]
  const float* W = (const float*)d_in[1];  // [512][4096]
  const float* U = (const float*)d_in[2];  // [1024][4096]
  const float* b = (const float*)d_in[3];  // [4096]
  float* out = (float*)d_out;              // [32][512][1024]

  char* ws = (char*)d_ws;
  size_t off = 0;
  __bf16* xb = (__bf16*)(ws + off); off += (size_t)16384 * 512 * 2;  // 16 MiB
  __bf16* Wt = (__bf16*)(ws + off); off += (size_t)4096 * 512 * 2;   //  4 MiB
  __bf16* Ut = (__bf16*)(ws + off); off += (size_t)4096 * 1024 * 2;  //  8 MiB
  __bf16* hbuf = (__bf16*)(ws + off); off += (size_t)2 * 32 * 1024 * 2;
  unsigned* bar = (unsigned*)(ws + off); off += 256;
  float* xz = (float*)(ws + off);   off += (size_t)512 * 32 * 4096 * 4; // 256 MiB

  {
    int n = 16384 * 512;
    cvt_f32_to_bf16_k<<<(n + 255) / 256, 256, 0, stream>>>(x, xb, n);
  }
  {
    int n = 512 * 4096;
    transpose_f32_to_bf16_k<<<(n + 255) / 256, 256, 0, stream>>>(W, Wt, 512, 4096);
  }
  {
    int n = 1024 * 4096;
    transpose_f32_to_bf16_k<<<(n + 255) / 256, 256, 0, stream>>>(U, Ut, 1024, 4096);
  }
  {
    int n = 2 * 32 * 1024;
    init_state_k<<<(n + 255) / 256, 256, 0, stream>>>(hbuf, bar, n);
  }

  // phase 1: 65536 waves = 8192 blocks of 8 waves
  lstm_xproj_wmma<<<8192, 256, 0, stream>>>(xb, Wt, b, xz);

  // phase 2: persistent, 16 co-resident blocks, device-wide barrier per step
  lstm_recurrent_wmma<<<16, 256, 0, stream>>>(xz, Ut, hbuf, out, bar, 16);
}
